// mrun_block_8022998909572
// MI455X (gfx1250) — matrix operand order repaired and checked by reading, not compiled, not run on hardware
//
#include <hip/hip_runtime.h>
#include <hip/hip_bf16.h>
#include <math.h>

// ---------------- problem constants ----------------
#define B_    2
#define S_    2048
#define E_    1024
#define H_    16
#define O_    32
#define FE_   4096
#define T_    (B_*S_)          // 4096 tokens
#define EPS_  1e-5f
#define CHUNK 64
#define NC    (S_/CHUNK)       // 32 chunks per sequence
#define LSTR  34               // LDS row stride (even -> 8B-aligned K pairs, conflict-free)

typedef __attribute__((ext_vector_type(2)))  float  v2f;
typedef __attribute__((ext_vector_type(8)))  float  v8f;
typedef __attribute__((ext_vector_type(16))) __bf16 v16bf;

struct U32x8 { uint4 a, b; };

__device__ __forceinline__ v16bf load_v16bf(const __bf16* p0, const __bf16* p1) {
  U32x8 u;
  u.a = *(const uint4*)p0;   // 8 bf16
  u.b = *(const uint4*)p1;   // 8 bf16
  return __builtin_bit_cast(v16bf, u);
}

__device__ __forceinline__ unsigned short f2bf_raw(float f) {
  unsigned u = __builtin_bit_cast(unsigned, f);
  u += 0x7FFFu + ((u >> 16) & 1u);          // round-to-nearest-even
  return (unsigned short)(u >> 16);
}
__device__ __forceinline__ __bf16 f2bf(float f) {
  unsigned short h = f2bf_raw(f);
  return __builtin_bit_cast(__bf16, h);
}
__device__ __forceinline__ unsigned pack_bf2(float lo, float hi) {
  return (unsigned)f2bf_raw(lo) | ((unsigned)f2bf_raw(hi) << 16);
}

// ============================================================
// LN stats (mean / rstd per token)
// ============================================================
__global__ __launch_bounds__(256) void ln_stats_kernel(const float* __restrict__ x,
                                                       float* __restrict__ mu,
                                                       float* __restrict__ rs) {
  const int t = blockIdx.x;
  const float* xr = x + (size_t)t * E_;
  float s = 0.f, s2 = 0.f;
  for (int i = threadIdx.x; i < E_; i += 256) { float v = xr[i]; s += v; s2 += v*v; }
  for (int off = 16; off > 0; off >>= 1) { s += __shfl_down(s, off); s2 += __shfl_down(s2, off); }
  __shared__ float as[8], as2[8];
  const int w = threadIdx.x >> 5;
  if ((threadIdx.x & 31) == 0) { as[w] = s; as2[w] = s2; }
  __syncthreads();
  if (threadIdx.x == 0) {
    float S = 0.f, S2 = 0.f;
    for (int i = 0; i < 8; ++i) { S += as[i]; S2 += as2[i]; }
    float m = S * (1.0f / E_);
    float v = S2 * (1.0f / E_) - m * m;
    mu[t] = m;
    rs[t] = rsqrtf(v + EPS_);
  }
}

// ============================================================
// LN2 + bf16 output (y = LN(x1)*g2), packed dword stores
// ============================================================
__global__ __launch_bounds__(256) void ln2_y_kernel(const float* __restrict__ x1,
                                                    const float* __restrict__ g2,
                                                    unsigned* __restrict__ y2 /* bf16 pairs */) {
  const int t = blockIdx.x;
  const float* xr = x1 + (size_t)t * E_;
  float s = 0.f, s2 = 0.f;
  for (int i = threadIdx.x; i < E_; i += 256) { float v = xr[i]; s += v; s2 += v*v; }
  for (int off = 16; off > 0; off >>= 1) { s += __shfl_down(s, off); s2 += __shfl_down(s2, off); }
  __shared__ float as[8], as2[8], smu, srs;
  const int w = threadIdx.x >> 5;
  if ((threadIdx.x & 31) == 0) { as[w] = s; as2[w] = s2; }
  __syncthreads();
  if (threadIdx.x == 0) {
    float S = 0.f, S2 = 0.f;
    for (int i = 0; i < 8; ++i) { S += as[i]; S2 += as2[i]; }
    float m = S * (1.0f / E_);
    float v = S2 * (1.0f / E_) - m * m;
    smu = m; srs = rsqrtf(v + EPS_);
  }
  __syncthreads();
  const float m = smu, r = srs;
  unsigned* yr = y2 + (size_t)t * (E_ / 2);
  for (int i = threadIdx.x; i < E_ / 2; i += 256) {
    const float a = (xr[2*i]   - m) * r * g2[2*i];
    const float b = (xr[2*i+1] - m) * r * g2[2*i+1];
    yr[i] = pack_bf2(a, b);
  }
}

// ============================================================
// fp32 -> bf16 weight conversion (packed dword stores)
// ============================================================
__global__ __launch_bounds__(256) void cvt_kernel(const float* __restrict__ src,
                                                  unsigned* __restrict__ dst, int npairs) {
  for (int i = blockIdx.x * 256 + threadIdx.x; i < npairs; i += gridDim.x * 256)
    dst[i] = pack_bf2(src[2*i], src[2*i+1]);
}

// ============================================================
// Wave-level 32x32 fp32 matmul step (P = P @ M) via WMMA f32 16x16x4.
// P is row-major [o][q], Mt is TRANSPOSED (Mt[n][k] = M[k][n]); both
// stride LSTR in LDS, so A and B operand pairs are aligned b64 loads.
// Single wave per block: LDS in-order per wave -> no barriers needed.
// ============================================================
__device__ __forceinline__ void wave_mm_step(float* P, const float* Mt, int lm, int hi) {
  v8f c00 = {}, c01 = {}, c10 = {}, c11 = {};
#pragma unroll
  for (int kk = 0; kk < 8; ++kk) {
    const int kb = kk * 4 + hi * 2;   // f32 A/B layout: lanes 16-31 hold K+2,K+3
    const v2f a0 = *(const v2f*)&P [ lm       * LSTR + kb];   // rows 0-15
    const v2f a1 = *(const v2f*)&P [(16 + lm) * LSTR + kb];   // rows 16-31
    const v2f b0 = *(const v2f*)&Mt[ lm       * LSTR + kb];   // cols 0-15
    const v2f b1 = *(const v2f*)&Mt[(16 + lm) * LSTR + kb];   // cols 16-31
    c00 = __builtin_amdgcn_wmma_f32_16x16x4_f32(false, a0, false, b0, (short)0, c00, false, false);
    c01 = __builtin_amdgcn_wmma_f32_16x16x4_f32(false, a0, false, b1, (short)0, c01, false, false);
    c10 = __builtin_amdgcn_wmma_f32_16x16x4_f32(false, a1, false, b0, (short)0, c10, false, false);
    c11 = __builtin_amdgcn_wmma_f32_16x16x4_f32(false, a1, false, b1, (short)0, c11, false, false);
  }
#pragma unroll
  for (int r = 0; r < 8; ++r) {        // C layout: vgpr r -> rows r (lanes<16) / r+8 (lanes>=16)
    P[(r + hi * 8) * LSTR + lm]            = c00[r];
    P[(r + hi * 8) * LSTR + 16 + lm]       = c01[r];
    P[(16 + r + hi * 8) * LSTR + lm]       = c10[r];
    P[(16 + r + hi * 8) * LSTR + 16 + lm]  = c11[r];
  }
}

// h[o][c] for this lane's row o of the transition matrix
__device__ __forceinline__ void ln_head(const float* xrow, float mu, float rs,
                                        const float* g1s, int lane, float& h0, float& h1) {
  h0 = (xrow[lane * 2 + 0] - mu) * rs * g1s[lane * 2 + 0];
  h1 = (xrow[lane * 2 + 1] - mu) * rs * g1s[lane * 2 + 1];
}

// M[o][q] = h0*Wup[q][0] + h1*Wup[q][1] + (o==q); store TRANSPOSED: Mt[q][o]
__device__ __forceinline__ void build_M_T(float* Mt, const float* xrow, float mu, float rs,
                                          const float* g1s, const float* wup, int lane) {
  float h0, h1; ln_head(xrow, mu, rs, g1s, lane, h0, h1);
#pragma unroll
  for (int q = 0; q < 32; ++q) {
    float m = h0 * wup[q * 2 + 0] + h1 * wup[q * 2 + 1];
    if (q == lane) m += 1.0f;
    Mt[q * LSTR + lane] = m;      // consecutive lanes -> consecutive addrs (conflict-free)
  }
}

// Same matrix stored row-major (used to seed P with the chunk's first matrix)
__device__ __forceinline__ void build_M_R(float* P, const float* xrow, float mu, float rs,
                                          const float* g1s, const float* wup, int lane) {
  float h0, h1; ln_head(xrow, mu, rs, g1s, lane, h0, h1);
#pragma unroll
  for (int q = 0; q < 32; ++q) {
    float m = h0 * wup[q * 2 + 0] + h1 * wup[q * 2 + 1];
    if (q == lane) m += 1.0f;
    P[lane * LSTR + q] = m;
  }
}

// ============================================================
// Scan pass 1: per-chunk total products  T = M_t0 @ ... @ M_t0+CHUNK-1
// grid = B*H*NC blocks of one wave (32 threads)
// ============================================================
__global__ __launch_bounds__(32) void scan_pass1(const float* __restrict__ x,
                                                 const float* __restrict__ mu1,
                                                 const float* __restrict__ rs1,
                                                 const float* __restrict__ Wup,
                                                 const float* __restrict__ g1,
                                                 float* __restrict__ Tprod) {
  __shared__ float sP[32 * LSTR], sMt[32 * LSTR], wup[64], g1s[64];
  const int lane = threadIdx.x;
  const int lm = lane & 15, hi = lane >> 4;
  const int id = blockIdx.x;
  const int chunk = id % NC;
  const int h = (id / NC) % H_;
  const int b = id / (NC * H_);

  wup[lane] = Wup[lane];           wup[32 + lane] = Wup[32 + lane];
  g1s[lane] = g1[h * 64 + lane];   g1s[32 + lane] = g1[h * 64 + 32 + lane];

  const int t0 = chunk * CHUNK;
  { // first matrix seeds P (row-major)
    const int tk = b * S_ + t0;
    build_M_R(sP, x + (size_t)tk * E_ + h * 64, mu1[tk], rs1[tk], g1s, wup, lane);
  }
  for (int t = t0 + 1; t < t0 + CHUNK; ++t) {
    const int tk = b * S_ + t;
    build_M_T(sMt, x + (size_t)tk * E_ + h * 64, mu1[tk], rs1[tk], g1s, wup, lane);
    wave_mm_step(sP, sMt, lm, hi);
  }
  float* T = Tprod + (size_t)id * 1024;
#pragma unroll
  for (int q = 0; q < 32; ++q) T[lane * 32 + q] = sP[lane * LSTR + q];
}

// ============================================================
// Inter-chunk exclusive prefix scan (per (b,h) chain, NC sequential steps)
// grid = B*H blocks of one wave
// ============================================================
__global__ __launch_bounds__(32) void chunk_scan(const float* __restrict__ Tprod,
                                                 float* __restrict__ Epre) {
  __shared__ float sP[32 * LSTR], sMt[32 * LSTR];
  const int lane = threadIdx.x;
  const int lm = lane & 15, hi = lane >> 4;
  const int id = blockIdx.x;  // b*H + h
#pragma unroll
  for (int q = 0; q < 32; ++q) sP[lane * LSTR + q] = (q == lane) ? 1.0f : 0.0f;
  for (int j = 0; j < NC; ++j) {
    const float* T = Tprod + ((size_t)id * NC + j) * 1024;   // row-major T[k][n]
    float* Eo      = Epre  + ((size_t)id * NC + j) * 1024;
#pragma unroll
    for (int q = 0; q < 32; ++q) {
      Eo[lane * 32 + q]    = sP[lane * LSTR + q];   // exclusive prefix (row-major)
      sMt[q * LSTR + lane] = T[lane * 32 + q];      // load transposed: Mt[n][k] = T[k][n]
    }
    wave_mm_step(sP, sMt, lm, hi);
  }
}

// ============================================================
// Scan pass 2: inclusive prefixes with carry-in, fused down-projection
// (bf16 output) and new_state store for last batch element.
// ============================================================
__global__ __launch_bounds__(32) void scan_pass2(const float* __restrict__ x,
                                                 const float* __restrict__ mu1,
                                                 const float* __restrict__ rs1,
                                                 const float* __restrict__ Wup,
                                                 const float* __restrict__ g1,
                                                 const float* __restrict__ Wdn,
                                                 const float* __restrict__ Epre,
                                                 unsigned* __restrict__ down2 /* bf16 pairs */,
                                                 float* __restrict__ newstate) {
  __shared__ float sP[32 * LSTR], sMt[32 * LSTR], wup[64], g1s[64], wd[64];
  const int lane = threadIdx.x;
  const int lm = lane & 15, hi = lane >> 4;
  const int id = blockIdx.x;
  const int chunk = id % NC;
  const int h = (id / NC) % H_;
  const int b = id / (NC * H_);

  wup[lane] = Wup[lane];           wup[32 + lane] = Wup[32 + lane];
  g1s[lane] = g1[h * 64 + lane];   g1s[32 + lane] = g1[h * 64 + 32 + lane];
  wd[lane]  = Wdn[lane];           wd[32 + lane]  = Wdn[32 + lane];

  const float* Ein = Epre + (size_t)id * 1024;   // carry-in prefix (exclusive, row-major)
#pragma unroll
  for (int q = 0; q < 32; ++q) sP[lane * LSTR + q] = Ein[lane * 32 + q];

  const int t0 = chunk * CHUNK;
  for (int t = t0; t < t0 + CHUNK; ++t) {
    const int tk = b * S_ + t;
    build_M_T(sMt, x + (size_t)tk * E_ + h * 64, mu1[tk], rs1[tk], g1s, wup, lane);
    wave_mm_step(sP, sMt, lm, hi);                // inclusive prefix P_t

    // down[o][c] = sum_q P[o][q] * Wdn[c][q]  (lane = o), packed bf16 pair store
    float d0 = 0.f, d1 = 0.f;
#pragma unroll
    for (int q = 0; q < 32; ++q) {
      const float p = sP[lane * LSTR + q];
      d0 += p * wd[q];
      d1 += p * wd[32 + q];
    }
    down2[(size_t)tk * (E_ / 2) + h * 32 + lane] = pack_bf2(d0, d1);

    if (b == B_ - 1) {                           // new_state = states[-1] : [S,H,O,O]
      float* ns = newstate + ((size_t)t * H_ + h) * 1024 + lane * 32;
#pragma unroll
      for (int q = 0; q < 32; ++q) ns[q] = sP[lane * LSTR + q];
    }
  }
}

// ============================================================
// bf16 GEMM (out = resid + A @ W^T), W stored [N,K] row-major.
// Each wave computes a 16x128 tile with v_wmma_f32_16x16x32_bf16,
// A/B loaded directly in WMMA register layout (2x b128 per operand).
// ============================================================
__global__ __launch_bounds__(256) void gemm_bf16_resid(const __bf16* __restrict__ A,
                                                       const __bf16* __restrict__ W,
                                                       const float* __restrict__ resid,
                                                       float* __restrict__ out,
                                                       int M, int N, int K) {
  const int wid  = (blockIdx.x * 256 + threadIdx.x) >> 5;
  const int lane = threadIdx.x & 31;
  const int lm = lane & 15, hi = lane >> 4;
  const int nt8 = N >> 7;                 // 128-wide column tiles
  const int mt = wid / nt8, nt = wid % nt8;
  const int m0 = mt * 16, n0 = nt * 128;

  v8f c[8] = {};
  const __bf16* arow = A + (size_t)(m0 + lm) * K;
  const int aoff0 = hi ? 16 : 0;          // contiguous k order used here (B matches): lanes 0-15 K 0-15, lanes 16-31 K 16-31
  const int aoff1 = hi ? 24 : 8;

  for (int k = 0; k < K; k += 32) {
    if (k + 64 < K) __builtin_prefetch(arow + k + 64, 0, 1);   // global_prefetch_b8
    const v16bf a = load_v16bf(arow + k + aoff0, arow + k + aoff1);
#pragma unroll
    for (int j = 0; j < 8; ++j) {
      const __bf16* brow = W + (size_t)(n0 + j * 16 + lm) * K + k + hi * 16;
      const v16bf bb = load_v16bf(brow, brow + 8);
      c[j] = __builtin_amdgcn_wmma_f32_16x16x32_bf16(false, a, false, bb, (short)0, c[j], false, false);
    }
  }
#pragma unroll
  for (int j = 0; j < 8; ++j)
#pragma unroll
    for (int r = 0; r < 8; ++r) {
      const size_t idx = (size_t)(m0 + r + hi * 8) * N + n0 + j * 16 + lm;
      out[idx] = resid[idx] + c[j][r];
    }
}

// Same GEMM, exact-gelu epilogue, bf16 output (MLP hidden layer)
__global__ __launch_bounds__(256) void gemm_bf16_gelu(const __bf16* __restrict__ A,
                                                      const __bf16* __restrict__ W,
                                                      __bf16* __restrict__ out,
                                                      int M, int N, int K) {
  const int wid  = (blockIdx.x * 256 + threadIdx.x) >> 5;
  const int lane = threadIdx.x & 31;
  const int lm = lane & 15, hi = lane >> 4;
  const int nt8 = N >> 7;
  const int mt = wid / nt8, nt = wid % nt8;
  const int m0 = mt * 16, n0 = nt * 128;

  v8f c[8] = {};
  const __bf16* arow = A + (size_t)(m0 + lm) * K;
  const int aoff0 = hi ? 16 : 0;
  const int aoff1 = hi ? 24 : 8;

  for (int k = 0; k < K; k += 32) {
    if (k + 64 < K) __builtin_prefetch(arow + k + 64, 0, 1);
    const v16bf a = load_v16bf(arow + k + aoff0, arow + k + aoff1);
#pragma unroll
    for (int j = 0; j < 8; ++j) {
      const __bf16* brow = W + (size_t)(n0 + j * 16 + lm) * K + k + hi * 16;
      const v16bf bb = load_v16bf(brow, brow + 8);
      c[j] = __builtin_amdgcn_wmma_f32_16x16x32_bf16(false, a, false, bb, (short)0, c[j], false, false);
    }
  }
#pragma unroll
  for (int j = 0; j < 8; ++j)
#pragma unroll
    for (int r = 0; r < 8; ++r) {
      const float v = c[j][r];
      const float g = 0.5f * v * (1.0f + erff(v * 0.70710678118654752f));  // exact gelu
      out[(size_t)(m0 + r + hi * 8) * N + n0 + j * 16 + lm] = f2bf(g);
    }
}

// ============================================================
// host launcher
// ============================================================
extern "C" void kernel_launch(void* const* d_in, const int* in_sizes, int n_in,
                              void* d_out, int out_size, void* d_ws, size_t ws_size,
                              hipStream_t stream) {
  const float* x    = (const float*)d_in[0];
  const float* Wup  = (const float*)d_in[1];
  const float* Wdn  = (const float*)d_in[2];
  const float* Wmru = (const float*)d_in[3];
  const float* g1   = (const float*)d_in[4];
  const float* g2   = (const float*)d_in[5];
  const float* W1   = (const float*)d_in[6];
  const float* W2   = (const float*)d_in[7];

  float* out_x     = (float*)d_out;
  float* out_state = out_x + (size_t)B_ * S_ * E_;

  char* w = (char*)d_ws;
  auto alloc = [&](size_t bytes) -> char* {
    char* p = w;
    w += (bytes + 255) & ~(size_t)255;
    return p;
  };
  float*  mu1   = (float*) alloc((size_t)T_ * 4);
  float*  rs1   = (float*) alloc((size_t)T_ * 4);
  float*  Tprod = (float*) alloc((size_t)B_ * H_ * NC * 1024 * 4);
  float*  Epre  = (float*) alloc((size_t)B_ * H_ * NC * 1024 * 4);
  __bf16* down  = (__bf16*)alloc((size_t)T_ * E_ * 2);
  float*  x1    = (float*) alloc((size_t)T_ * E_ * 4);
  __bf16* ybf   = (__bf16*)alloc((size_t)T_ * E_ * 2);
  __bf16* gmid  = (__bf16*)alloc((size_t)T_ * FE_ * 2);
  __bf16* mrub  = (__bf16*)alloc((size_t)E_ * E_ * 2);
  __bf16* w1b   = (__bf16*)alloc((size_t)FE_ * E_ * 2);
  __bf16* w2b   = (__bf16*)alloc((size_t)E_ * FE_ * 2);

  // Phase 0: LN1 stats + bf16 weight conversion
  ln_stats_kernel<<<T_, 256, 0, stream>>>(x, mu1, rs1);
  cvt_kernel<<<1024, 256, 0, stream>>>(Wmru, (unsigned*)mrub, E_ * E_ / 2);
  cvt_kernel<<<4096, 256, 0, stream>>>(W1, (unsigned*)w1b, FE_ * E_ / 2);
  cvt_kernel<<<4096, 256, 0, stream>>>(W2, (unsigned*)w2b, E_ * FE_ / 2);

  // Phase 1-3: chunked prefix matrix-product scan (fp32 WMMA), fused LN1/up/down
  scan_pass1<<<B_ * H_ * NC, 32, 0, stream>>>(x, mu1, rs1, Wup, g1, Tprod);
  chunk_scan<<<B_ * H_, 32, 0, stream>>>(Tprod, Epre);
  scan_pass2<<<B_ * H_ * NC, 32, 0, stream>>>(x, mu1, rs1, Wup, g1, Wdn, Epre,
                                              (unsigned*)down, out_state);

  // Phase 4: mru_out GEMM + residual  (x1 = x + down @ Wmru^T)
  gemm_bf16_resid<<<(T_ / 16) * (E_ / 128) / 8, 256, 0, stream>>>(down, mrub, x, x1, T_, E_, E_);

  // Phase 5: LN2 -> bf16 y
  ln2_y_kernel<<<T_, 256, 0, stream>>>(x1, g2, (unsigned*)ybf);

  // Phase 6-7: MLP (gelu hidden, residual out)
  gemm_bf16_gelu <<<(T_ / 16) * (FE_ / 128) / 8, 256, 0, stream>>>(ybf, w1b, gmid, T_, FE_, E_);
  gemm_bf16_resid<<<(T_ / 16) * (E_ / 128) / 8, 256, 0, stream>>>(gmid, w2b, x1, out_x, T_, E_, FE_);
}
